// Radiomic_mamba_encoder_15899968930096
// MI455X (gfx1250) — matrix operand order repaired and checked by reading, not compiled, not run on hardware
//
#include <hip/hip_runtime.h>
#include <hip/hip_bf16.h>

// ---------------------------------------------------------------------------
// Types for CDNA5 WMMA (bf16 -> f32, 16x16x32)
// ---------------------------------------------------------------------------
typedef __bf16 bf16;
typedef __attribute__((ext_vector_type(16))) __bf16 v16bf;
typedef __attribute__((ext_vector_type(8)))  __bf16 v8bf;
typedef __attribute__((ext_vector_type(8)))  float  v8f;

static __device__ __forceinline__ bf16 f32_to_bf16(float f) {
    unsigned u = __builtin_bit_cast(unsigned, f);
    u += 0x7FFFu + ((u >> 16) & 1u);          // round-to-nearest-even
    unsigned short h = (unsigned short)(u >> 16);
    return __builtin_bit_cast(bf16, h);
}

// ---------------------------------------------------------------------------
// fp32 -> bf16 conversion with K padding (rows x Kin  ->  rows x Kout)
// ---------------------------------------------------------------------------
__global__ void cvt_bf16_pad(const float* __restrict__ src, bf16* __restrict__ dst,
                             int rows, int Kin, int Kout) {
    int idx = blockIdx.x * blockDim.x + threadIdx.x;
    if (idx >= rows * Kout) return;
    int r = idx / Kout;
    int k = idx - r * Kout;
    float v = (k < Kin) ? src[(size_t)r * Kin + k] : 0.0f;
    dst[idx] = f32_to_bf16(v);
}

// ---------------------------------------------------------------------------
// C[m,n] = sum_k A[m,k] * B[n,k]   (A: MxK bf16 row-major, B: NxK bf16 row-major)
// One wave computes a 32x64 output strip: 2 A fragments x 4 B fragments ->
// 8 independent WMMAs per 32-deep K step (hides XDL latency, 0.75KB L2/WMMA).
// K mult of 32; M mult of 32; N mult of 64.
// ---------------------------------------------------------------------------
__global__ void gemm_bf16_wmma(const bf16* __restrict__ A, const bf16* __restrict__ B,
                               float* __restrict__ C, int M, int N, int K) {
    const int wave    = (blockIdx.x * blockDim.x + threadIdx.x) >> 5;
    const int lane    = threadIdx.x & 31;
    const int tilesN4 = N >> 6;                 // groups of four 16-wide N tiles
    const int tm  = wave / tilesN4;             // 32-row strip index
    const int tn4 = wave % tilesN4;
    if (tm >= (M >> 5)) return;                 // wave-uniform guard

    const int laneLo  = lane & 15;
    const int laneHi  = lane >> 4;
    const int row0    = tm * 32 + laneLo;       // A rows for this lane
    const int row1    = row0 + 16;
    const int colBase = tn4 * 64 + laneLo;      // first B row (output column)

    v8f acc00 = {}, acc01 = {}, acc02 = {}, acc03 = {};
    v8f acc10 = {}, acc11 = {}, acc12 = {}, acc13 = {};
    const bf16* arow0 = A + (size_t)row0 * K;
    const bf16* arow1 = A + (size_t)row1 * K;

    for (int k = 0; k < K; k += 32) {
        // A fragment, contiguous K order (matches B):
        //   lanes 0-15 : elems[0..15]=K0..15
        //   lanes 16-31: elems[0..15]=K16..31
        const int ka = k + (laneHi << 4);
        v8bf a0lo = *(const v8bf*)(arow0 + ka);
        v8bf a0hi = *(const v8bf*)(arow0 + ka + 8);
        v8bf a1lo = *(const v8bf*)(arow1 + ka);
        v8bf a1hi = *(const v8bf*)(arow1 + ka + 8);
        v16bf a0, a1;
#pragma unroll
        for (int i = 0; i < 8; ++i) {
            a0[i] = a0lo[i]; a0[i + 8] = a0hi[i];
            a1[i] = a1lo[i]; a1[i + 8] = a1hi[i];
        }

        // B fragment: lane = column; lanes 0-15 hold K k..k+15, lanes 16-31 hold K k+16..k+31
        const int kb = k + (laneHi << 4);
        v16bf b0 = *(const v16bf*)(B + (size_t)(colBase +  0) * K + kb);
        v16bf b1 = *(const v16bf*)(B + (size_t)(colBase + 16) * K + kb);
        v16bf b2 = *(const v16bf*)(B + (size_t)(colBase + 32) * K + kb);
        v16bf b3 = *(const v16bf*)(B + (size_t)(colBase + 48) * K + kb);

        acc00 = __builtin_amdgcn_wmma_f32_16x16x32_bf16(false, a0, false, b0, (short)0, acc00, false, false);
        acc01 = __builtin_amdgcn_wmma_f32_16x16x32_bf16(false, a0, false, b1, (short)0, acc01, false, false);
        acc02 = __builtin_amdgcn_wmma_f32_16x16x32_bf16(false, a0, false, b2, (short)0, acc02, false, false);
        acc03 = __builtin_amdgcn_wmma_f32_16x16x32_bf16(false, a0, false, b3, (short)0, acc03, false, false);
        acc10 = __builtin_amdgcn_wmma_f32_16x16x32_bf16(false, a1, false, b0, (short)0, acc10, false, false);
        acc11 = __builtin_amdgcn_wmma_f32_16x16x32_bf16(false, a1, false, b1, (short)0, acc11, false, false);
        acc12 = __builtin_amdgcn_wmma_f32_16x16x32_bf16(false, a1, false, b2, (short)0, acc12, false, false);
        acc13 = __builtin_amdgcn_wmma_f32_16x16x32_bf16(false, a1, false, b3, (short)0, acc13, false, false);
    }

    // C/D layout: VGPR i -> M = i + 8*laneHi ; lane%16 -> N
    const int r0 = tm * 32 + (laneHi << 3);
#pragma unroll
    for (int i = 0; i < 8; ++i) {
        float* crow = C + (size_t)(r0 + i) * N + colBase;
        crow[0]  = acc00[i];
        crow[16] = acc01[i];
        crow[32] = acc02[i];
        crow[48] = acc03[i];
        float* drow = crow + (size_t)16 * N;
        drow[0]  = acc10[i];
        drow[16] = acc11[i];
        drow[32] = acc12[i];
        drow[48] = acc13[i];
    }
}

// ---------------------------------------------------------------------------
// Batchnorm stats, two deterministic passes for parallelism over rows.
// Pass 1: per (slice, column) partial sums.  Pass 2: finalize mean/rstd.
// ---------------------------------------------------------------------------
__global__ void bn_partial(const float* __restrict__ X, float* __restrict__ psum,
                           float* __restrict__ psq, int rows, int cols, int slices) {
    int c = blockIdx.x * blockDim.x + threadIdx.x;
    int p = blockIdx.y;
    if (c >= cols) return;
    int span = rows / slices;
    int r0 = p * span, r1 = r0 + span;
    float s = 0.f, s2 = 0.f;
    for (int r = r0; r < r1; ++r) {
        float v = X[(size_t)r * cols + c];
        s += v; s2 += v * v;
    }
    psum[(size_t)p * cols + c] = s;
    psq [(size_t)p * cols + c] = s2;
}

__global__ void bn_finalize(const float* __restrict__ psum, const float* __restrict__ psq,
                            float* __restrict__ mean, float* __restrict__ rstd,
                            int rows, int cols, int slices) {
    int c = blockIdx.x * blockDim.x + threadIdx.x;
    if (c >= cols) return;
    float s = 0.f, s2 = 0.f;
    for (int p = 0; p < slices; ++p) {
        s  += psum[(size_t)p * cols + c];
        s2 += psq [(size_t)p * cols + c];
    }
    float m   = s / (float)rows;
    float var = s2 / (float)rows - m * m;
    mean[c] = m;
    rstd[c] = rsqrtf(var + 1e-5f);
}

// ---------------------------------------------------------------------------
// batchnorm + relu + bf16 convert (elementwise)
// ---------------------------------------------------------------------------
__global__ void bn_relu_cvt(const float* __restrict__ X, const float* __restrict__ mean,
                            const float* __restrict__ rstd, const float* __restrict__ g,
                            const float* __restrict__ b, bf16* __restrict__ Y,
                            int rows, int cols) {
    int idx = blockIdx.x * blockDim.x + threadIdx.x;
    if (idx >= rows * cols) return;
    int c = idx % cols;
    float v = g[c] * (X[idx] - mean[c]) * rstd[c] + b[c];
    Y[idx] = f32_to_bf16(fmaxf(v, 0.0f));
}

// ---------------------------------------------------------------------------
// Mamba scan: one wave32 per batch row. lane = (e = lane/16, n = lane%16).
// Reads raw GEMM1 output and applies batchnorm(g1,b1,mean1,rstd1) inline.
// Writes bf16 output (the GEMM2 activation).  L must be a multiple of 32.
// ---------------------------------------------------------------------------
__global__ void mamba_scan(const float* __restrict__ C1,
                           const float* __restrict__ mean1, const float* __restrict__ rstd1,
                           const float* __restrict__ g1, const float* __restrict__ b1,
                           const float* __restrict__ W_in, const float* __restrict__ conv_w,
                           const float* __restrict__ conv_b, const float* __restrict__ W_xproj,
                           const float* __restrict__ W_dt, const float* __restrict__ b_dt,
                           const float* __restrict__ A_log, const float* __restrict__ Dp,
                           const float* __restrict__ W_out,
                           bf16* __restrict__ out, int Bsz, int L) {
    const int lane = threadIdx.x & 31;
    const int wave = (blockIdx.x * blockDim.x + threadIdx.x) >> 5;
    if (wave >= Bsz) return;
    const int e = lane >> 4;       // inner channel (0..1)
    const int n = lane & 15;       // state index  (0..15)

    // per-lane constants
    const float win_x = W_in[e];
    const float win_z = W_in[2 + e];
    const float cw0 = conv_w[e * 4 + 0], cw1 = conv_w[e * 4 + 1];
    const float cw2 = conv_w[e * 4 + 2], cw3 = conv_w[e * 4 + 3];
    const float cb  = conv_b[e];
    const float wx00 = W_xproj[0], wx01 = W_xproj[1];          // dt row
    const float wB0 = W_xproj[(1 + n) * 2 + 0],  wB1 = W_xproj[(1 + n) * 2 + 1];
    const float wC0 = W_xproj[(17 + n) * 2 + 0], wC1 = W_xproj[(17 + n) * 2 + 1];
    const float wdt = W_dt[e];
    const float bdt = b_dt[e];
    const float Aen = -__expf(A_log[e * 16 + n]);
    const float De  = Dp[e];
    const float wo  = W_out[e];

    const float* crow = C1 + (size_t)wave * L;
    bf16* orow = out + (size_t)wave * L;

    float h = 0.f;                       // SSM state for (e,n)
    float h0 = 0.f, h1 = 0.f, h2 = 0.f;  // conv history of x_pre (per channel e)

    for (int t0 = 0; t0 < L; t0 += 32) {
        const int tl = t0 + lane;        // coalesced load of 32 timesteps
        const float uval = g1[tl] * (crow[tl] - mean1[tl]) * rstd1[tl] + b1[tl];
        if (t0 + 32 < L) {               // hide the next chunk behind the serial chain
            __builtin_prefetch(crow + tl + 32, 0, 0);
        }
        float accum = 0.f;

        for (int s = 0; s < 32; ++s) {
            const float u = __shfl(uval, s, 32);
            const float xpre = win_x * u;
            const float zv   = win_z * u;
            // causal depthwise conv (cross-correlation, left pad 3)
            const float xc = cw0 * h0 + cw1 * h1 + cw2 * h2 + cw3 * xpre + cb;
            h0 = h1; h1 = h2; h2 = xpre;
            const float xs = xc / (1.f + __expf(-xc));          // silu
            // exchange to get both channels' x in every lane
            const float xo = __shfl_xor(xs, 16, 32);
            const float x0 = (e == 0) ? xs : xo;
            const float x1 = (e == 0) ? xo : xs;
            const float dt_in = wx00 * x0 + wx01 * x1;
            const float pre   = wdt * dt_in + bdt;
            const float dt    = (pre > 20.f) ? pre : log1pf(__expf(pre));  // softplus
            const float Bn = wB0 * x0 + wB1 * x1;
            const float Cn = wC0 * x0 + wC1 * x1;
            const float dA = __expf(dt * Aen);
            h = h * dA + dt * Bn * xs;
            float r = h * Cn;                                   // reduce over n (16 lanes)
            r += __shfl_xor(r, 1, 32);
            r += __shfl_xor(r, 2, 32);
            r += __shfl_xor(r, 4, 32);
            r += __shfl_xor(r, 8, 32);
            float y = r + De * xs;
            y *= zv / (1.f + __expf(-zv));                      // * silu(z)
            const float oc = y * wo;                            // W_out contraction
            const float ot = oc + __shfl_xor(oc, 16, 32);
            if (lane == s) accum = ot;
        }
        orow[t0 + lane] = f32_to_bf16(accum);                   // coalesced store
    }
}

// ---------------------------------------------------------------------------
// Launch
// ---------------------------------------------------------------------------
extern "C" void kernel_launch(void* const* d_in, const int* in_sizes, int n_in,
                              void* d_out, int out_size, void* d_ws, size_t ws_size,
                              hipStream_t stream) {
    (void)in_sizes; (void)n_in; (void)out_size; (void)ws_size;

    const float* x      = (const float*)d_in[0];
    const float* W1     = (const float*)d_in[1];
    const float* g1     = (const float*)d_in[2];
    const float* b1     = (const float*)d_in[3];
    const float* W_in   = (const float*)d_in[4];
    const float* conv_w = (const float*)d_in[5];
    const float* conv_b = (const float*)d_in[6];
    const float* W_xprj = (const float*)d_in[7];
    const float* W_dt   = (const float*)d_in[8];
    const float* b_dt   = (const float*)d_in[9];
    const float* A_log  = (const float*)d_in[10];
    const float* Dp     = (const float*)d_in[11];
    const float* W_out  = (const float*)d_in[12];
    const float* W2     = (const float*)d_in[13];
    const float* g2     = (const float*)d_in[14];
    const float* b2     = (const float*)d_in[15];
    const float* W3     = (const float*)d_in[16];
    float* out = (float*)d_out;

    const int B   = 1024;
    const int K1  = 1775, K1P = 1792;   // pad to mult of 32
    const int E1  = 2048;               // GEMM1 N / mamba L
    const int N2  = 512;
    const int N3  = 256;
    const int SL  = 16;                 // bn row slices

    // workspace carve-up (aligned to 256B)
    char* ws = (char*)d_ws;
    size_t o = 0;
    auto alloc = [&](size_t bytes) { char* p = ws + o; o += (bytes + 255) & ~(size_t)255; return p; };
    bf16*  xb    = (bf16*)alloc((size_t)B  * K1P * 2);
    bf16*  w1b   = (bf16*)alloc((size_t)E1 * K1P * 2);
    bf16*  w2b   = (bf16*)alloc((size_t)N2 * E1  * 2);
    bf16*  w3b   = (bf16*)alloc((size_t)N3 * N2  * 2);
    float* c1    = (float*)alloc((size_t)B * E1 * 4);
    bf16*  h2b   = (bf16*) alloc((size_t)B * E1 * 2);
    float* c2    = (float*)alloc((size_t)B * N2 * 4);
    bf16*  h3b   = (bf16*) alloc((size_t)B * N2 * 2);
    float* mean1 = (float*)alloc(E1 * 4);
    float* rstd1 = (float*)alloc(E1 * 4);
    float* mean2 = (float*)alloc(N2 * 4);
    float* rstd2 = (float*)alloc(N2 * 4);
    float* psum  = (float*)alloc((size_t)SL * E1 * 4);
    float* psq   = (float*)alloc((size_t)SL * E1 * 4);

    const int T = 256;
    auto nb = [](int total, int t) { return (total + t - 1) / t; };

    // Stage A: bf16 conversions (with K padding)
    cvt_bf16_pad<<<nb(B  * K1P, T), T, 0, stream>>>(x,  xb,  B,  K1, K1P);
    cvt_bf16_pad<<<nb(E1 * K1P, T), T, 0, stream>>>(W1, w1b, E1, K1, K1P);
    cvt_bf16_pad<<<nb(N2 * E1,  T), T, 0, stream>>>(W2, w2b, N2, E1, E1);
    cvt_bf16_pad<<<nb(N3 * N2,  T), T, 0, stream>>>(W3, w3b, N3, N2, N2);

    // Stage B: GEMM1 (1024x1792)x(1792x2048) -> c1
    {
        int waves = (B / 32) * (E1 / 64);
        gemm_bf16_wmma<<<waves / 8, T, 0, stream>>>(xb, w1b, c1, B, E1, K1P);
    }

    // Stage C: batchnorm stats over batch for c1 columns (two-pass, deterministic)
    bn_partial<<<dim3(nb(E1, T), SL), T, 0, stream>>>(c1, psum, psq, B, E1, SL);
    bn_finalize<<<nb(E1, T), T, 0, stream>>>(psum, psq, mean1, rstd1, B, E1, SL);

    // Stage D: mamba scan (batchnorm applied inline), writes bf16 h2b
    mamba_scan<<<(B * 32) / T, T, 0, stream>>>(c1, mean1, rstd1, g1, b1,
                                               W_in, conv_w, conv_b, W_xprj,
                                               W_dt, b_dt, A_log, Dp, W_out,
                                               h2b, B, E1);

    // Stage E: GEMM2 (1024x2048)x(2048x512) -> c2
    {
        int waves = (B / 32) * (N2 / 64);
        gemm_bf16_wmma<<<waves / 8, T, 0, stream>>>(h2b, w2b, c2, B, N2, E1);
    }

    // Stage F: batchnorm + relu -> bf16 h3b
    bn_partial<<<dim3(nb(N2, T), SL), T, 0, stream>>>(c2, psum, psq, B, N2, SL);
    bn_finalize<<<nb(N2, T), T, 0, stream>>>(psum, psq, mean2, rstd2, B, N2, SL);
    bn_relu_cvt<<<nb(B * N2, T), T, 0, stream>>>(c2, mean2, rstd2, g2, b2, h3b, B, N2);

    // Stage G: GEMM3 (1024x512)x(512x256) -> out
    {
        int waves = (B / 32) * (N3 / 64);
        gemm_bf16_wmma<<<waves / 8, T, 0, stream>>>(h3b, w3b, out, B, N3, N2);
    }
}
